// LSTMAttentionDecoder_21277267984620
// MI455X (gfx1250) — matrix operand order repaired and checked by reading, not compiled, not run on hardware
//
#include <hip/hip_runtime.h>
#include <math.h>

// Problem dims (match reference)
#define I_DIM 256
#define H_DIM 512
#define C_DIM 1024
#define G_DIM 2048   // 4*H
#define TY    32
#define TX    128
#define B_DIM 32

typedef __attribute__((ext_vector_type(16))) __bf16 v16bf;
typedef __attribute__((ext_vector_type(8)))  __bf16 v8bf;
typedef __attribute__((ext_vector_type(8)))  float  v8f;

__device__ __forceinline__ float sigmoidf_(float x) {
  return 1.0f / (1.0f + __expf(-x));
}

// ---------------------------------------------------------------------------
// bf16 fragment loads (16-byte aligned global_load_b128 pairs).
// Layouts used here (contiguous k order, same for A and B):
//   A 16x32 bf16: lane low = M row;   e   -> K=hi*16+e
//   B 32x16 bf16: lane low = N col;   e   -> K=hi*16+e
// ---------------------------------------------------------------------------
__device__ __forceinline__ v16bf load_a_frag(const __bf16* __restrict__ arow,
                                             int k0, int hi) {
  v16bf a;
  ((v8bf*)&a)[0] = *(const v8bf*)(arow + k0 + hi * 16);
  ((v8bf*)&a)[1] = *(const v8bf*)(arow + k0 + hi * 16 + 8);
  return a;
}
__device__ __forceinline__ v16bf load_b_frag(const __bf16* __restrict__ wrow,
                                             int k) {
  v16bf b;
  ((v8bf*)&b)[0] = *(const v8bf*)(wrow + k);
  ((v8bf*)&b)[1] = *(const v8bf*)(wrow + k + 8);
  return b;
}

// ---------------------------------------------------------------------------
// Wave-level K loop over a 16x64 strip, software-pipelined:
// loads for slab k+32 are issued before the 4 WMMAs of slab k, so ~10 b128
// loads are in flight behind every WMMA group (partial s_wait_loadcnt).
// ---------------------------------------------------------------------------
__device__ __forceinline__ void kloop_strip(const __bf16* __restrict__ A, int lda,
                                            const __bf16* __restrict__ W, int ldw,
                                            int K, int m0, int n0, int low, int hi,
                                            v8f acc[4]) {
  const __bf16* arow = A + (size_t)(m0 + low) * (size_t)lda;
  const __bf16* w0 = W + (size_t)(n0 + low) * (size_t)ldw;
  const __bf16* w1 = w0 + (size_t)16 * (size_t)ldw;
  const __bf16* w2 = w1 + (size_t)16 * (size_t)ldw;
  const __bf16* w3 = w2 + (size_t)16 * (size_t)ldw;

  // prologue: fragments for k0 = 0
  v16bf a  = load_a_frag(arow, 0, hi);
  v16bf b0 = load_b_frag(w0, hi * 16);
  v16bf b1 = load_b_frag(w1, hi * 16);
  v16bf b2 = load_b_frag(w2, hi * 16);
  v16bf b3 = load_b_frag(w3, hi * 16);

  for (int k0 = 0; k0 < K; k0 += 32) {
    v16bf ca = a, cb0 = b0, cb1 = b1, cb2 = b2, cb3 = b3;
    const int kn = k0 + 32;
    if (kn < K) {                                // wave-uniform
      a  = load_a_frag(arow, kn, hi);
      b0 = load_b_frag(w0, kn + hi * 16);
      b1 = load_b_frag(w1, kn + hi * 16);
      b2 = load_b_frag(w2, kn + hi * 16);
      b3 = load_b_frag(w3, kn + hi * 16);
    }
    acc[0] = __builtin_amdgcn_wmma_f32_16x16x32_bf16(false, ca, false, cb0,
                                                     (short)0, acc[0], false, false);
    acc[1] = __builtin_amdgcn_wmma_f32_16x16x32_bf16(false, ca, false, cb1,
                                                     (short)0, acc[1], false, false);
    acc[2] = __builtin_amdgcn_wmma_f32_16x16x32_bf16(false, ca, false, cb2,
                                                     (short)0, acc[2], false, false);
    acc[3] = __builtin_amdgcn_wmma_f32_16x16x32_bf16(false, ca, false, cb3,
                                                     (short)0, acc[3], false, false);
  }
}

// ---------------------------------------------------------------------------
// Generic  D = [bias] + [Cadd] + A@W^T (+ A2@W2^T)
// A/W pre-converted bf16; one wave per 16x64 strip. N must be multiple of 64.
// Wave-uniform control flow => EXEC all-1s at every WMMA.
// ---------------------------------------------------------------------------
__global__ __launch_bounds__(256)
void gemm_bf16_kernel(const __bf16* __restrict__ A,  int lda,
                      const __bf16* __restrict__ W,  int ldw, int K,
                      const __bf16* __restrict__ A2, int lda2,
                      const __bf16* __restrict__ W2, int ldw2, int K2,
                      const float* __restrict__ bias,
                      const float* __restrict__ Cadd, int ldc,
                      float* __restrict__ D, int ldd, int M, int N) {
  const int strip   = blockIdx.x * (blockDim.x >> 5) + (threadIdx.x >> 5);
  const int stripsN = N >> 6;
  const int total   = (M >> 4) * stripsN;
  if (strip >= total) return;                    // wave-uniform exit
  const int m0 = (strip / stripsN) << 4;
  const int n0 = (strip % stripsN) << 6;
  const int lane = threadIdx.x & 31;
  const int low  = lane & 15;
  const int hi   = lane >> 4;

  v8f acc[4];
#pragma unroll
  for (int j = 0; j < 4; ++j) {
    const int n = n0 + 16 * j + low;
    const float bv = bias ? bias[n] : 0.0f;
#pragma unroll
    for (int e = 0; e < 8; ++e) {
      float v = bv;
      if (Cadd) v += Cadd[(size_t)(m0 + hi * 8 + e) * (size_t)ldc + n];
      acc[j][e] = v;
    }
  }

  kloop_strip(A, lda, W, ldw, K, m0, n0, low, hi, acc);
  if (A2) kloop_strip(A2, lda2, W2, ldw2, K2, m0, n0, low, hi, acc);

#pragma unroll
  for (int j = 0; j < 4; ++j)
#pragma unroll
    for (int e = 0; e < 8; ++e)                  // D: m = hi*8+e, n = low
      D[(size_t)(m0 + hi * 8 + e) * (size_t)ldd + n0 + 16 * j + low] = acc[j][e];
}

// ---------------------------------------------------------------------------
// One-time fp32 -> bf16 conversion (weights / context / y_emb)
// ---------------------------------------------------------------------------
__global__ void cvt_bf16_kernel(const float* __restrict__ src,
                                __bf16* __restrict__ dst, int n) {
  int i = blockIdx.x * blockDim.x + threadIdx.x;
  if (i < n) dst[i] = (__bf16)src[i];
}

// ---------------------------------------------------------------------------
// State init: h_prev = init_h (fp32 + bf16 mirror), c_prev = init_c
// ---------------------------------------------------------------------------
__global__ void init_state_kernel(const float* __restrict__ ih,
                                  const float* __restrict__ ic,
                                  float* __restrict__ h_prev,
                                  float* __restrict__ c_prev,
                                  __bf16* __restrict__ h_prev_bf) {
  int idx = blockIdx.x * blockDim.x + threadIdx.x;
  if (idx < B_DIM * H_DIM) {
    float h = ih[idx];
    h_prev[idx] = h;
    h_prev_bf[idx] = (__bf16)h;
    c_prev[idx] = ic[idx];
  }
}

// ---------------------------------------------------------------------------
// LSTM cell 1
// ---------------------------------------------------------------------------
__global__ void cell1_kernel(const float* __restrict__ pre1,
                             const float* __restrict__ h_prev,
                             const float* __restrict__ c_prev,
                             const float* __restrict__ ym_t,
                             float* __restrict__ h1, float* __restrict__ c1,
                             __bf16* __restrict__ h1_bf) {
  int idx = blockIdx.x * blockDim.x + threadIdx.x;
  if (idx >= B_DIM * H_DIM) return;
  int b = idx / H_DIM, h = idx - b * H_DIM;
  const float* row = pre1 + (size_t)b * G_DIM;
  float i1 = row[h], f1 = row[H_DIM + h], o1 = row[2 * H_DIM + h],
        g1 = row[3 * H_DIM + h];
  float c = sigmoidf_(f1) * c_prev[idx] + sigmoidf_(i1) * tanhf(g1);
  float hh = sigmoidf_(o1) * tanhf(c);
  float ym = ym_t[b];
  float hv = ym * hh + (1.0f - ym) * h_prev[idx];
  h1[idx] = hv;
  h1_bf[idx] = (__bf16)hv;
  c1[idx] = c;                                   // carry c1 unmasked (as ref)
}

// ---------------------------------------------------------------------------
// Attention energies: one wave per (t_src, b); pctx is L2-resident (16MB<<192MB)
// ---------------------------------------------------------------------------
__global__ __launch_bounds__(256)
void score_kernel(const float* __restrict__ pctx,
                  const float* __restrict__ hproj,
                  const float* __restrict__ U_att,
                  const float* __restrict__ x_mask,
                  float* __restrict__ score) {
  int p = blockIdx.x * (blockDim.x >> 5) + (threadIdx.x >> 5);
  if (p >= TX * B_DIM) return;
  int lane = threadIdx.x & 31;
  int b = p & (B_DIM - 1);
  const float* prow = pctx + (size_t)p * C_DIM;
  const float* hrow = hproj + (size_t)b * C_DIM;
  float acc = 0.0f;
  for (int c = lane; c < C_DIM; c += 32) {
    __builtin_prefetch(prow + c + 256, 0, 1);    // global_prefetch_b8
    acc += U_att[c] * tanhf(prow[c] + hrow[c]);
  }
#pragma unroll
  for (int off = 16; off; off >>= 1) acc += __shfl_xor(acc, off, 32);
  if (lane == 0) score[p] = acc * x_mask[p];
}

// ---------------------------------------------------------------------------
// Per-batch softmax over TX + context reduction; writes atts (fp32 out) and
// bf16 atted for the pre2 WMMA GEMM.
// ---------------------------------------------------------------------------
__global__ __launch_bounds__(256)
void softmax_atted_kernel(const float* __restrict__ score,
                          const float* __restrict__ x_mask,
                          const float* __restrict__ context,
                          __bf16* __restrict__ atted_bf,
                          float* __restrict__ atts_out) {
  __shared__ float wt[TX];
  __shared__ float invsum;
  const int b = blockIdx.x, tid = threadIdx.x;
  if (tid < TX) wt[tid] = score[tid * B_DIM + b];
  __syncthreads();
  if (tid == 0) {
    float m = wt[0];
    for (int i = 1; i < TX; ++i) m = fmaxf(m, wt[i]);
    float s = 0.0f;
    for (int i = 0; i < TX; ++i) {
      float e = __expf(wt[i] - m) * x_mask[i * B_DIM + b];
      wt[i] = e; s += e;
    }
    invsum = 1.0f / s;
  }
  __syncthreads();
  const float inv = invsum;
  for (int c = tid; c < C_DIM; c += blockDim.x) {
    float acc = 0.0f;
#pragma unroll 4
    for (int ts = 0; ts < TX; ++ts)
      acc += wt[ts] * context[((size_t)ts * B_DIM + b) * C_DIM + c];
    acc *= inv;
    atted_bf[(size_t)b * C_DIM + c] = (__bf16)acc;
    atts_out[(size_t)b * C_DIM + c] = acc;
  }
}

// ---------------------------------------------------------------------------
// LSTM cell 2 + output writes + state update (fp32 + bf16 mirror of h)
// ---------------------------------------------------------------------------
__global__ void cell2_kernel(const float* __restrict__ pre2,
                             const float* __restrict__ h1,
                             const float* __restrict__ c1,
                             const float* __restrict__ ym_t,
                             float* __restrict__ h_prev,
                             float* __restrict__ c_prev,
                             __bf16* __restrict__ h_prev_bf,
                             float* __restrict__ hs_out,
                             float* __restrict__ cs_out) {
  int idx = blockIdx.x * blockDim.x + threadIdx.x;
  if (idx >= B_DIM * H_DIM) return;
  int b = idx / H_DIM, h = idx - b * H_DIM;
  const float* row = pre2 + (size_t)b * G_DIM;
  float i2 = row[h], f2 = row[H_DIM + h], o2 = row[2 * H_DIM + h],
        g2 = row[3 * H_DIM + h];
  float c = sigmoidf_(f2) * c1[idx] + sigmoidf_(i2) * tanhf(g2);
  float hh = sigmoidf_(o2) * tanhf(c);
  float ym = ym_t[b];
  float h2 = ym * hh + (1.0f - ym) * h1[idx];
  h_prev[idx] = h2;  h_prev_bf[idx] = (__bf16)h2;
  c_prev[idx] = c;
  hs_out[idx] = h2;  cs_out[idx] = c;
}

// ---------------------------------------------------------------------------
// Host-side orchestration (all on `stream`, graph-capture safe)
// ---------------------------------------------------------------------------
extern "C" void kernel_launch(void* const* d_in, const int* in_sizes, int n_in,
                              void* d_out, int out_size, void* d_ws, size_t ws_size,
                              hipStream_t stream) {
  (void)in_sizes; (void)n_in; (void)out_size; (void)ws_size;
  const float* y_emb   = (const float*)d_in[0];   // [TY,B,I]
  const float* context = (const float*)d_in[1];   // [TX,B,C]
  const float* init_h  = (const float*)d_in[2];   // [B,H]
  const float* init_c  = (const float*)d_in[3];   // [B,H]
  const float* x_mask  = (const float*)d_in[4];   // [TX,B,1]
  const float* y_mask  = (const float*)d_in[5];   // [TY,B,1]
  const float* Wm      = (const float*)d_in[6];   // [4H,I]
  const float* Um      = (const float*)d_in[7];   // [4H,H]
  const float* bv      = (const float*)d_in[8];   // [4H]
  const float* Wx      = (const float*)d_in[9];   // [4H,C]
  const float* Ux      = (const float*)d_in[10];  // [4H,H]
  const float* bx      = (const float*)d_in[11];  // [4H]
  const float* Wc_att  = (const float*)d_in[12];  // [C,C]
  const float* b_att   = (const float*)d_in[13];  // [C]
  const float* W_comb  = (const float*)d_in[14];  // [C,H]
  const float* U_att   = (const float*)d_in[15];  // [1,C]

  float* out    = (float*)d_out;
  float* hs_out = out;                                   // [TY,B,H]
  float* cs_out = out + (size_t)TY * B_DIM * H_DIM;      // [TY,B,H]
  float* atts   = out + (size_t)2 * TY * B_DIM * H_DIM;  // [TY,B,C]

  // Workspace layout (~46 MB; fp32 intermediates + one-time bf16 mirrors)
  char* w = (char*)d_ws;
  float* pctx   = (float*)w;  w += (size_t)TX * B_DIM * C_DIM * 4;   // 16 MB
  float* xbuf   = (float*)w;  w += (size_t)TY * B_DIM * G_DIM * 4;   //  8 MB
  float* pre1   = (float*)w;  w += (size_t)B_DIM * G_DIM * 4;
  float* pre2   = (float*)w;  w += (size_t)B_DIM * G_DIM * 4;
  float* hproj  = (float*)w;  w += (size_t)B_DIM * C_DIM * 4;
  float* score  = (float*)w;  w += (size_t)TX * B_DIM * 4;
  float* h_prev = (float*)w;  w += (size_t)B_DIM * H_DIM * 4;
  float* c_prev = (float*)w;  w += (size_t)B_DIM * H_DIM * 4;
  float* h1     = (float*)w;  w += (size_t)B_DIM * H_DIM * 4;
  float* c1     = (float*)w;  w += (size_t)B_DIM * H_DIM * 4;
  __bf16* ctx_bf   = (__bf16*)w;  w += (size_t)TX * B_DIM * C_DIM * 2; // 8 MB
  __bf16* yemb_bf  = (__bf16*)w;  w += (size_t)TY * B_DIM * I_DIM * 2;
  __bf16* Wm_bf    = (__bf16*)w;  w += (size_t)G_DIM * I_DIM * 2;
  __bf16* Um_bf    = (__bf16*)w;  w += (size_t)G_DIM * H_DIM * 2;
  __bf16* Ux_bf    = (__bf16*)w;  w += (size_t)G_DIM * H_DIM * 2;
  __bf16* Wx_bf    = (__bf16*)w;  w += (size_t)G_DIM * C_DIM * 2;
  __bf16* Wc_bf    = (__bf16*)w;  w += (size_t)C_DIM * C_DIM * 2;
  __bf16* Wcomb_bf = (__bf16*)w;  w += (size_t)C_DIM * H_DIM * 2;
  __bf16* hprev_bf = (__bf16*)w;  w += (size_t)B_DIM * H_DIM * 2;
  __bf16* h1_bf    = (__bf16*)w;  w += (size_t)B_DIM * H_DIM * 2;
  __bf16* atted_bf = (__bf16*)w;  w += (size_t)B_DIM * C_DIM * 2;

  // 0) one-time conversions + state init
  auto cvt = [&](const float* s, __bf16* d, int n) {
    cvt_bf16_kernel<<<(n + 255) / 256, 256, 0, stream>>>(s, d, n);
  };
  cvt(context, ctx_bf,   TX * B_DIM * C_DIM);
  cvt(y_emb,   yemb_bf,  TY * B_DIM * I_DIM);
  cvt(Wm,      Wm_bf,    G_DIM * I_DIM);
  cvt(Um,      Um_bf,    G_DIM * H_DIM);
  cvt(Ux,      Ux_bf,    G_DIM * H_DIM);
  cvt(Wx,      Wx_bf,    G_DIM * C_DIM);
  cvt(Wc_att,  Wc_bf,    C_DIM * C_DIM);
  cvt(W_comb,  Wcomb_bf, C_DIM * H_DIM);
  init_state_kernel<<<64, 256, 0, stream>>>(init_h, init_c, h_prev, c_prev,
                                            hprev_bf);

  // 1) pctx = context @ Wc_att^T + b_att   (M=4096, N=1024) — 4096 strips
  gemm_bf16_kernel<<<512, 256, 0, stream>>>(
      ctx_bf, C_DIM, Wc_bf, C_DIM, C_DIM,
      nullptr, 0, nullptr, 0, 0,
      b_att, nullptr, 0, pctx, C_DIM, TX * B_DIM, C_DIM);

  // 2) x = y_emb @ W^T + b   (M=1024, N=2048) — 2048 strips
  gemm_bf16_kernel<<<256, 256, 0, stream>>>(
      yemb_bf, I_DIM, Wm_bf, I_DIM, I_DIM,
      nullptr, 0, nullptr, 0, 0,
      bv, nullptr, 0, xbuf, G_DIM, TY * B_DIM, G_DIM);

  // 3) sequential decode over TY steps
  for (int t = 0; t < TY; ++t) {
    const float* xt  = xbuf   + (size_t)t * B_DIM * G_DIM;
    const float* ymt = y_mask + (size_t)t * B_DIM;

    // pre1 = x_t + h_prev @ U^T   (M=32, N=2048, K=512) — 64 strips
    gemm_bf16_kernel<<<8, 256, 0, stream>>>(
        hprev_bf, H_DIM, Um_bf, H_DIM, H_DIM,
        nullptr, 0, nullptr, 0, 0,
        nullptr, xt, G_DIM, pre1, G_DIM, B_DIM, G_DIM);

    cell1_kernel<<<64, 256, 0, stream>>>(pre1, h_prev, c_prev, ymt, h1, c1,
                                         h1_bf);

    // hproj = h1 @ W_comb_att^T   (M=32, N=1024, K=512) — 32 strips
    gemm_bf16_kernel<<<4, 256, 0, stream>>>(
        h1_bf, H_DIM, Wcomb_bf, H_DIM, H_DIM,
        nullptr, 0, nullptr, 0, 0,
        nullptr, nullptr, 0, hproj, C_DIM, B_DIM, C_DIM);

    // attention energies (4096 waves) + softmax/context reduction
    score_kernel<<<512, 256, 0, stream>>>(pctx, hproj, U_att, x_mask, score);
    softmax_atted_kernel<<<B_DIM, 256, 0, stream>>>(
        score, x_mask, context, atted_bf, atts + (size_t)t * B_DIM * C_DIM);

    // pre2 = bx + h1 @ Ux^T + atted @ Wx^T  (dual product) — 64 strips
    gemm_bf16_kernel<<<8, 256, 0, stream>>>(
        h1_bf, H_DIM, Ux_bf, H_DIM, H_DIM,
        atted_bf, C_DIM, Wx_bf, C_DIM, C_DIM,
        bx, nullptr, 0, pre2, G_DIM, B_DIM, G_DIM);

    cell2_kernel<<<64, 256, 0, stream>>>(
        pre2, h1, c1, ymt, h_prev, c_prev, hprev_bf,
        hs_out + (size_t)t * B_DIM * H_DIM,
        cs_out + (size_t)t * B_DIM * H_DIM);
  }
}
